// GCT_70987219468387
// MI455X (gfx1250) — matrix operand order repaired and checked by reading, not compiled, not run on hardware
//
#include <hip/hip_runtime.h>

#define B_ 8
#define C_ 256
#define HID_ 256
#define H_ 128
#define W_ 128
#define N_ (H_*W_)
#define S_ 256
#define NH_ 4
#define DH_ 64
#define EPS_ 1e-8f

typedef __attribute__((ext_vector_type(16))) _Float16 v16h;
typedef __attribute__((ext_vector_type(8)))  float    v8f;

// ---------------------------------------------------------------- utilities
__global__ void k_zero(float* p, long long n) {
  long long i = (long long)blockIdx.x * blockDim.x + threadIdx.x;
  long long st = (long long)gridDim.x * blockDim.x;
  for (; i < n; i += st) p[i] = 0.0f;
}

// --------------------------------------------------- superpixel color stats
__global__ void k_colsum(const float* __restrict__ img, const int* __restrict__ seg,
                         float* __restrict__ colsum) {
  int b = blockIdx.z;
  int p = blockIdx.x * 256 + threadIdx.x;
  int s = seg[(size_t)b * N_ + p] & (S_ - 1);
  float r  = img[((size_t)b * 3 + 0) * N_ + p];
  float g  = img[((size_t)b * 3 + 1) * N_ + p];
  float bl = img[((size_t)b * 3 + 2) * N_ + p];
  float* cs = colsum + ((size_t)b * S_ + s) * 4;
  atomicAdd(cs + 0, r);
  atomicAdd(cs + 1, g);
  atomicAdd(cs + 2, bl);
  atomicAdd(cs + 3, 1.0f);
}

// ------------------------------------------- per-pixel cosine sim + ||O||^2
__global__ void k_simnf(const float* __restrict__ img, const int* __restrict__ seg,
                        const float* __restrict__ colsum,
                        float* __restrict__ sim, float* __restrict__ nf) {
  int b = blockIdx.z;
  int p = blockIdx.x * 256 + threadIdx.x;
  int s = seg[(size_t)b * N_ + p] & (S_ - 1);
  const float* cs = colsum + ((size_t)b * S_ + s) * 4;
  float inv = 1.0f / (cs[3] + EPS_);
  float mr = cs[0] * inv, mg = cs[1] * inv, mb = cs[2] * inv;
  float r  = img[((size_t)b * 3 + 0) * N_ + p];
  float g  = img[((size_t)b * 3 + 1) * N_ + p];
  float bl = img[((size_t)b * 3 + 2) * N_ + p];
  float na = fmaxf(sqrtf(r * r + g * g + bl * bl), EPS_);
  float nb = fmaxf(sqrtf(mr * mr + mg * mg + mb * mb), EPS_);
  float sv = (r * mr + g * mg + bl * mb) / (na * nb);
  sv = fminf(fmaxf(sv, 0.0f), 1.0f);
  sim[(size_t)b * N_ + p] = sv;
  atomicAdd(&nf[b * S_ + s], sv * sv);
}

// ----------------- heavy pass: node accumulation from x (run-compressed)
// Each lane owns 8 contiguous pixels; a wave covers a contiguous 1KB span of
// x (coalesced). An atomic is emitted only when the segment id changes
// within the run -> ~1 atomic per 8 elements for coherent labels.
__global__ void k_noderaw(const float* __restrict__ x, const int* __restrict__ seg,
                          const float* __restrict__ sim, float* __restrict__ nraw) {
  int b = blockIdx.z, c = blockIdx.y;
  int p0 = (blockIdx.x * 256 + threadIdx.x) * 8;
  const float* xp = x   + ((size_t)b * C_ + c) * N_ + p0;
  const int*   sp = seg + (size_t)b * N_ + p0;
  const float* sm = sim + (size_t)b * N_ + p0;
  float* nb = nraw + (size_t)b * S_ * C_ + c;
  __builtin_prefetch(xp + 2048, 0, 0);        // global_prefetch_b8
  int cur = sp[0] & (S_ - 1);
  float acc = 0.0f;
#pragma unroll
  for (int i = 0; i < 8; ++i) {
    int s = sp[i] & (S_ - 1);
    float v = sm[i] * xp[i];
    if (s != cur) { atomicAdd(nb + (size_t)cur * C_, acc); acc = 0.0f; cur = s; }
    acc += v;
  }
  atomicAdd(nb + (size_t)cur * C_, acc);
}

__global__ void k_nodediv(float* __restrict__ nraw, const float* __restrict__ nf) {
  long long i = (long long)blockIdx.x * 256 + threadIdx.x;   // B*S*C
  long long sc = i / C_;
  nraw[i] /= (nf[sc] + EPS_);
}

// ------------------------------------------------------- adjacency (S x S)
__global__ void k_adj(const int* __restrict__ seg, float* __restrict__ Ab) {
  int b = blockIdx.z;
  int p = blockIdx.x * 256 + threadIdx.x;
  int py = p / W_, px = p % W_;
  int u = seg[(size_t)b * N_ + p] & (S_ - 1);
  float* A = Ab + (size_t)b * S_ * S_;
  if (px + 1 < W_) {
    int v = seg[(size_t)b * N_ + p + 1] & (S_ - 1);
    if (u != v) { A[u * S_ + v] = 1.0f; A[v * S_ + u] = 1.0f; }
  }
  if (py + 1 < H_) {
    int v = seg[(size_t)b * N_ + p + W_] & (S_ - 1);
    if (u != v) { A[u * S_ + v] = 1.0f; A[v * S_ + u] = 1.0f; }
  }
}

__global__ void k_deg(const float* __restrict__ Ab, float* __restrict__ dis) {
  int b = blockIdx.z, s = blockIdx.x, t = threadIdx.x;
  __shared__ float red[256];
  red[t] = Ab[((size_t)b * S_ + s) * S_ + t];
  __syncthreads();
  for (int o = 128; o > 0; o >>= 1) { if (t < o) red[t] += red[t + o]; __syncthreads(); }
  if (t == 0) dis[b * S_ + s] = rsqrtf(red[0] + 1.0f);   // +1 self loop
}

__global__ void k_annorm(float* __restrict__ Ab, const float* __restrict__ dis) {
  long long i = (long long)blockIdx.x * 256 + threadIdx.x;   // B*S*S
  int b = (int)(i / (S_ * S_));
  int rem = (int)(i % (S_ * S_));
  int u = rem / S_, v = rem % S_;
  float a = (u == v) ? 1.0f : Ab[i];
  Ab[i] = a * dis[b * S_ + u] * dis[b * S_ + v];
}

// ------------------------------------------------------------- WMMA GEMM
// C[M,N] = act( A[M,K] @ B + bias + res ), B either [N,K] (BT=1) or [K,N].
// One wave computes a 16x64 strip: 4 accumulators reuse one A fragment.
struct GemmP {
  const float* A;   long long sA; int lda;
  const float* Bm;  long long sB; int ldb;
  const float* bias;
  const float* res; long long sR; int ldr;
  float*       C;   long long sC; int ldc;
  int M, N, K;
  int relu;
};

template <int BT>
__global__ void __launch_bounds__(128) k_gemm(GemmP g) {
  int lane = threadIdx.x;                       // 0..31 (wave32)
  int m0 = (blockIdx.y * blockDim.y + threadIdx.y) * 16;
  int n0 = blockIdx.x * 64;
  if (m0 >= g.M || n0 >= g.N) return;           // wave-uniform: EXEC stays all-ones
  int bz = blockIdx.z;
  const float* A  = g.A  + (long long)bz * g.sA;
  const float* Bm = g.Bm + (long long)bz * g.sB;
  int half = lane >> 4, lo = lane & 15;

  v8f acc0 = {}, acc1 = {}, acc2 = {}, acc3 = {};
  for (int k0 = 0; k0 < g.K; k0 += 32) {
    // A fragment (16x32 f16): lanes 0-15 hold K 0-15, 16-31 hold K 16-31
    v16h af;
    const float* ar = A + (size_t)(m0 + lo) * g.lda + k0;
#pragma unroll
    for (int j = 0; j < 8; ++j) {
      int kb = j * 2 + half * 16;
      af[2 * j]     = (_Float16)ar[kb];
      af[2 * j + 1] = (_Float16)ar[kb + 1];
    }
    // 4 B fragments (32x16 f16 each): lanes 0-15 hold K 0-15, 16-31 hold K 16-31
    v8f* accs[4] = {&acc0, &acc1, &acc2, &acc3};
#pragma unroll
    for (int t = 0; t < 4; ++t) {
      v16h bf;
      if (BT) {   // B stored [N,K] row-major: element(k,n)=Bm[n*ldb+k]
        const float* br = Bm + (size_t)(n0 + t * 16 + lo) * g.ldb + k0 + half * 16;
#pragma unroll
        for (int j = 0; j < 8; ++j) {
          bf[2 * j]     = (_Float16)br[2 * j];
          bf[2 * j + 1] = (_Float16)br[2 * j + 1];
        }
      } else {    // B stored [K,N] row-major: element(k,n)=Bm[k*ldb+n]
        const float* bc = Bm + (size_t)(k0 + half * 16) * g.ldb + (n0 + t * 16 + lo);
#pragma unroll
        for (int j = 0; j < 8; ++j) {
          bf[2 * j]     = (_Float16)bc[(size_t)(2 * j) * g.ldb];
          bf[2 * j + 1] = (_Float16)bc[(size_t)(2 * j + 1) * g.ldb];
        }
      }
      *accs[t] = __builtin_amdgcn_wmma_f32_16x16x32_f16(
          false, af, false, bf, (short)0, *accs[t], false, false);
    }
  }

  float* Cc = g.C + (long long)bz * g.sC;
  const float* R = g.res ? g.res + (long long)bz * g.sR : nullptr;
  v8f* accs[4] = {&acc0, &acc1, &acc2, &acc3};
#pragma unroll
  for (int t = 0; t < 4; ++t) {
    int n = n0 + t * 16 + lo;
    float bia = g.bias ? g.bias[n] : 0.0f;
#pragma unroll
    for (int j = 0; j < 8; ++j) {
      int m = m0 + half * 8 + j;
      float v = (*accs[t])[j] + bia;
      if (R) v += R[(size_t)m * g.ldr + n];
      if (g.relu) v = fmaxf(v, 0.0f);
      Cc[(size_t)m * g.ldc + n] = v;
    }
  }
}

// ----------------------------------------------------------- softmax rows
__global__ void k_softmax(float* __restrict__ scores) {
  int q = blockIdx.x, h = blockIdx.y, b = blockIdx.z, t = threadIdx.x;
  float* row = scores + (((size_t)b * NH_ + h) * S_ + q) * S_;
  __shared__ float red[256];
  float v = row[t] * 0.125f;                 // 1/sqrt(dh), dh=64
  red[t] = v; __syncthreads();
  for (int o = 128; o > 0; o >>= 1) { if (t < o) red[t] = fmaxf(red[t], red[t + o]); __syncthreads(); }
  float mx = red[0]; __syncthreads();
  float e = __expf(v - mx);
  red[t] = e; __syncthreads();
  for (int o = 128; o > 0; o >>= 1) { if (t < o) red[t] += red[t + o]; __syncthreads(); }
  row[t] = e / red[0];
}

// -------------------------------------------------------------- LayerNorm
__global__ void k_ln(const float* __restrict__ in, const float* __restrict__ gam,
                     const float* __restrict__ bet, float* __restrict__ out) {
  int s = blockIdx.x, b = blockIdx.z, t = threadIdx.x;
  const float* row = in + ((size_t)b * S_ + s) * HID_;
  __shared__ float red[256];
  float x = row[t];
  red[t] = x; __syncthreads();
  for (int o = 128; o > 0; o >>= 1) { if (t < o) red[t] += red[t + o]; __syncthreads(); }
  float m = red[0] * (1.0f / HID_); __syncthreads();
  float d = x - m;
  red[t] = d * d; __syncthreads();
  for (int o = 128; o > 0; o >>= 1) { if (t < o) red[t] += red[t + o]; __syncthreads(); }
  float var = red[0] * (1.0f / HID_);
  out[((size_t)b * S_ + s) * HID_ + t] = d * rsqrtf(var + 1e-5f) * gam[t] + bet[t];
}

// ---------------------------------------------------- final pixel scatter
__global__ void k_scatter(const int* __restrict__ seg, const float* __restrict__ sim,
                          const float* __restrict__ y, float* __restrict__ out) {
  int b = blockIdx.z, c = blockIdx.y;
  int p = blockIdx.x * 256 + threadIdx.x;
  int s = seg[(size_t)b * N_ + p] & (S_ - 1);
  out[((size_t)b * HID_ + c) * N_ + p] =
      sim[(size_t)b * N_ + p] * y[((size_t)b * S_ + s) * HID_ + c];
}

// ------------------------------------------------------------------ host
extern "C" void kernel_launch(void* const* d_in, const int* in_sizes, int n_in,
                              void* d_out, int out_size, void* d_ws, size_t ws_size,
                              hipStream_t stream) {
  (void)in_sizes; (void)n_in; (void)out_size; (void)ws_size;
  const float* x   = (const float*)d_in[0];
  const float* img = (const float*)d_in[1];
  const int*   seg = (const int*)d_in[2];
  const float* Wp  = (const float*)d_in[3];  const float* bp  = (const float*)d_in[4];
  const float* Wg  = (const float*)d_in[5];  const float* bg  = (const float*)d_in[6];
  const float* ipw = (const float*)d_in[7];  const float* ipb = (const float*)d_in[8];
  const float* ow  = (const float*)d_in[9];  const float* ob  = (const float*)d_in[10];
  const float* l1g = (const float*)d_in[11]; const float* l1b = (const float*)d_in[12];
  const float* l2g = (const float*)d_in[13]; const float* l2b = (const float*)d_in[14];
  const float* f1w = (const float*)d_in[15]; const float* f1b = (const float*)d_in[16];
  const float* f2w = (const float*)d_in[17]; const float* f2b = (const float*)d_in[18];
  float* out = (float*)d_out;
  float* ws  = (float*)d_ws;

  // workspace layout (floats); total ~10.1M floats (~40.4 MB)
  size_t o_colsum = 0;
  size_t o_nf   = o_colsum + (size_t)B_ * S_ * 4;
  size_t o_sim  = o_nf   + (size_t)B_ * S_;
  size_t o_dis  = o_sim  + (size_t)B_ * N_;
  size_t o_nraw = o_dis  + (size_t)B_ * S_;
  size_t o_node = o_nraw + (size_t)B_ * S_ * C_;
  size_t o_h    = o_node + (size_t)B_ * S_ * HID_;
  size_t o_g    = o_h    + (size_t)B_ * S_ * HID_;
  size_t o_A    = o_g    + (size_t)B_ * S_ * HID_;
  size_t o_qkv  = o_A    + (size_t)B_ * S_ * S_;
  size_t o_sc   = o_qkv  + (size_t)B_ * S_ * 3 * HID_;
  size_t o_ao   = o_sc   + (size_t)B_ * NH_ * S_ * S_;
  size_t o_t1   = o_ao   + (size_t)B_ * S_ * HID_;
  size_t o_y1   = o_t1   + (size_t)B_ * S_ * HID_;
  size_t o_ff   = o_y1   + (size_t)B_ * S_ * HID_;
  size_t o_t2   = o_ff   + (size_t)B_ * S_ * 2 * HID_;
  size_t o_y2   = o_t2   + (size_t)B_ * S_ * HID_;
  size_t total  = o_y2   + (size_t)B_ * S_ * HID_;

  dim3 blk256(256);
  dim3 gpix(N_ / 256, 1, B_);

  k_zero<<<2048, blk256, 0, stream>>>(ws, (long long)total);
  k_colsum<<<gpix, blk256, 0, stream>>>(img, seg, ws + o_colsum);
  k_simnf<<<gpix, blk256, 0, stream>>>(img, seg, ws + o_colsum, ws + o_sim, ws + o_nf);
  k_noderaw<<<dim3(N_ / 2048, C_, B_), blk256, 0, stream>>>(x, seg, ws + o_sim, ws + o_nraw);
  k_nodediv<<<(B_ * S_ * C_) / 256, blk256, 0, stream>>>(ws + o_nraw, ws + o_nf);
  k_adj<<<gpix, blk256, 0, stream>>>(seg, ws + o_A);
  k_deg<<<dim3(S_, 1, B_), blk256, 0, stream>>>(ws + o_A, ws + o_dis);
  k_annorm<<<(B_ * S_ * S_) / 256, blk256, 0, stream>>>(ws + o_A, ws + o_dis);

  auto gemm = [&](GemmP p, int bT, int nbatch) {
    dim3 blk(32, 4);
    dim3 grd((p.N + 63) / 64, (p.M + 63) / 64, nbatch);
    if (bT) k_gemm<1><<<grd, blk, 0, stream>>>(p);
    else    k_gemm<0><<<grd, blk, 0, stream>>>(p);
  };
  long long SH = (long long)S_ * HID_;

  // node = noderaw @ Wp^T + bp
  gemm({ws + o_nraw, (long long)S_ * C_, C_,  Wp, 0, C_,  bp,
        nullptr, 0, 0,  ws + o_node, SH, HID_,  S_, HID_, C_, 0}, 1, B_);
  // h = node @ Wg^T
  gemm({ws + o_node, SH, HID_,  Wg, 0, HID_,  nullptr,
        nullptr, 0, 0,  ws + o_h, SH, HID_,  S_, HID_, HID_, 0}, 1, B_);
  // g = relu(An @ h + bg)
  gemm({ws + o_A, (long long)S_ * S_, S_,  ws + o_h, SH, HID_,  bg,
        nullptr, 0, 0,  ws + o_g, SH, HID_,  S_, HID_, S_, 1}, 0, B_);
  // qkv = g @ in_proj_w^T + in_proj_b
  gemm({ws + o_g, SH, HID_,  ipw, 0, HID_,  ipb,
        nullptr, 0, 0,  ws + o_qkv, (long long)S_ * 3 * HID_, 3 * HID_,
        S_, 3 * HID_, HID_, 0}, 1, B_);
  // per-head scores = q @ k^T, then softmax, then ao = att @ v
  for (int h = 0; h < NH_; ++h) {
    gemm({ws + o_qkv + (size_t)h * DH_, (long long)S_ * 3 * HID_, 3 * HID_,
          ws + o_qkv + HID_ + (size_t)h * DH_, (long long)S_ * 3 * HID_, 3 * HID_,
          nullptr, nullptr, 0, 0,
          ws + o_sc + (size_t)h * S_ * S_, (long long)NH_ * S_ * S_, S_,
          S_, S_, DH_, 0}, 1, B_);
  }
  k_softmax<<<dim3(S_, NH_, B_), blk256, 0, stream>>>(ws + o_sc);
  for (int h = 0; h < NH_; ++h) {
    gemm({ws + o_sc + (size_t)h * S_ * S_, (long long)NH_ * S_ * S_, S_,
          ws + o_qkv + 2 * HID_ + (size_t)h * DH_, (long long)S_ * 3 * HID_, 3 * HID_,
          nullptr, nullptr, 0, 0,
          ws + o_ao + (size_t)h * DH_, SH, HID_,
          S_, DH_, S_, 0}, 0, B_);
  }
  // t1 = ao @ out_w^T + out_b + g ; y1 = LN1(t1)
  gemm({ws + o_ao, SH, HID_,  ow, 0, HID_,  ob,
        ws + o_g, SH, HID_,  ws + o_t1, SH, HID_,  S_, HID_, HID_, 0}, 1, B_);
  k_ln<<<dim3(S_, 1, B_), blk256, 0, stream>>>(ws + o_t1, l1g, l1b, ws + o_y1);
  // ff = relu(y1 @ ff1^T + b1) ; t2 = ff @ ff2^T + b2 + y1 ; y2 = LN2(t2)
  gemm({ws + o_y1, SH, HID_,  f1w, 0, HID_,  f1b,
        nullptr, 0, 0,  ws + o_ff, (long long)S_ * 2 * HID_, 2 * HID_,
        S_, 2 * HID_, HID_, 1}, 1, B_);
  gemm({ws + o_ff, (long long)S_ * 2 * HID_, 2 * HID_,  f2w, 0, 2 * HID_,  f2b,
        ws + o_y1, SH, HID_,  ws + o_t2, SH, HID_,  S_, HID_, 2 * HID_, 0}, 1, B_);
  k_ln<<<dim3(S_, 1, B_), blk256, 0, stream>>>(ws + o_t2, l2g, l2b, ws + o_y2);

  // out[b,c,:,:] = sim * y2[seg]
  k_scatter<<<dim3(N_ / 256, HID_, B_), blk256, 0, stream>>>(seg, ws + o_sim, ws + o_y2, out);
  (void)out;
}
